// E1CellSimple_2147483648056
// MI455X (gfx1250) — matrix operand order repaired and checked by reading, not compiled, not run on hardware
//
#include <hip/hip_runtime.h>

#define B 64
#define T 2048
#define D 512
#define LDA 520   // padded LDS row stride (bf16 elems) -> conflict-free 16-row b128 reads
#define BB 16     // batch rows per scan block

typedef __attribute__((ext_vector_type(16))) __bf16 v16bf;
typedef __attribute__((ext_vector_type(8)))  float  v8f;

union BfVec { uint4 q[2]; v16bf v; };

__device__ __forceinline__ unsigned short f2bf(float f) {
  unsigned u = __builtin_bit_cast(unsigned, f);
  unsigned r = u + 0x7FFFu + ((u >> 16) & 1u);   // round-to-nearest-even
  return (unsigned short)(r >> 16);
}
__device__ __forceinline__ float bf2f(unsigned short b) {
  return __builtin_bit_cast(float, (unsigned)b << 16);
}

// ---------------- Phase 0: weights fp32 -> bf16 (L2-resident thereafter) ----
__global__ void cvt_weights(const float* __restrict__ Wh, const float* __restrict__ Wx,
                            const float* __restrict__ Wg,
                            unsigned short* __restrict__ WhB, unsigned short* __restrict__ WxB,
                            unsigned short* __restrict__ WgB) {
  int i = blockIdx.x * blockDim.x + threadIdx.x;
  if (i < D * D) {
    WhB[i] = f2bf(Wh[i]);
    WxB[i] = f2bf(Wx[i]);
    WgB[i] = f2bf(Wg[i]);
  }
}

// ---------------- Phase 1: xh = x@Wx^T + bh ; gate = sigmoid(x@Wg^T + bg) ---
// Block: 256 thr (8 waves). Block tile: M=16 (flat b*T+t rows), full N=512,
// both weight matrices (A tile in LDS reused by all 8 waves / 8 WMMAs per K-step).
// Outputs stored bf16, TIME-MAJOR [T,B,D] for streaming reads in the scan.
__global__ __launch_bounds__(256)
void proj_gemm(const float* __restrict__ x,
               const unsigned short* __restrict__ WxB,
               const unsigned short* __restrict__ WgB,
               const float* __restrict__ bh, const float* __restrict__ bg,
               unsigned short* __restrict__ xhB, unsigned short* __restrict__ gateB) {
  __shared__ __align__(16) unsigned short sA[16 * LDA];
  const int tid = threadIdx.x;
  const long mbase = (long)blockIdx.x * 16;

  // cooperative load + convert A tile [16 x 512] fp32 -> bf16 LDS
  for (int i = tid; i < 16 * (D / 4); i += 256) {
    int r  = i / (D / 4);
    int c4 = i % (D / 4);
    const float4 v = *(const float4*)(x + (mbase + r) * D + c4 * 4);
    unsigned lo = (unsigned)f2bf(v.x) | ((unsigned)f2bf(v.y) << 16);
    unsigned hi = (unsigned)f2bf(v.z) | ((unsigned)f2bf(v.w) << 16);
    *(uint2*)(sA + r * LDA + c4 * 4) = make_uint2(lo, hi);
  }
  __syncthreads();

  const int lane = tid & 31;
  const int wv   = tid >> 5;
  const int m16  = lane & 15;
  const int half = lane >> 4;
  const int nbase = wv * 64;           // 4 N-tiles per wave, for both weights

  v8f zero = {0.f, 0.f, 0.f, 0.f, 0.f, 0.f, 0.f, 0.f};
  v8f acc[8];
#pragma unroll
  for (int i = 0; i < 8; ++i) acc[i] = zero;

  for (int k0 = 0; k0 < D; k0 += 32) {
    BfVec a;                            // A operand, contiguous k = 16*half + s (matches B)
    a.q[0] = *(const uint4*)(sA + m16 * LDA + k0 + half * 16);
    a.q[1] = *(const uint4*)(sA + m16 * LDA + k0 + half * 16 + 8);
#pragma unroll
    for (int t4 = 0; t4 < 4; ++t4) {
      const int n = nbase + t4 * 16 + m16;
      const unsigned short* px = WxB + n * D + k0 + half * 16;
      const unsigned short* pg = WgB + n * D + k0 + half * 16;
      BfVec bx, bgv;
      bx.q[0]  = *(const uint4*)(px);
      bx.q[1]  = *(const uint4*)(px + 8);
      bgv.q[0] = *(const uint4*)(pg);
      bgv.q[1] = *(const uint4*)(pg + 8);
      acc[t4]     = __builtin_amdgcn_wmma_f32_16x16x32_bf16(false, a.v, false, bx.v,
                                                            (short)0, acc[t4], false, false);
      acc[4 + t4] = __builtin_amdgcn_wmma_f32_16x16x32_bf16(false, a.v, false, bgv.v,
                                                            (short)0, acc[4 + t4], false, false);
    }
  }

#pragma unroll
  for (int t4 = 0; t4 < 4; ++t4) {
    const int n = nbase + t4 * 16 + m16;
    const float vbh = bh[n];
    const float vbg = bg[n];
#pragma unroll
    for (int j = 0; j < 8; ++j) {
      const long row = mbase + j + 8 * half;   // flat b*T + t
      const long b   = row >> 11;              // / T
      const long t   = row & (T - 1);          // % T
      const long idx = (t * B + b) * D + n;    // time-major [T,B,D]
      xhB[idx] = f2bf(acc[t4][j] + vbh);
      const float gv = acc[4 + t4][j] + vbg;
      gateB[idx] = f2bf(1.0f / (1.0f + __expf(-gv)));
    }
  }
}

// ---------------- Phase 2: sequential scan, batch-partitioned (no grid sync) -
// 4 blocks x 512 thr (16 waves). Each block owns 16 batch rows; h lives in LDS
// (fp32 + bf16 WMMA-layout copy). W_h streamed from L2 each step; wave w owns
// output dims [w*32, w*32+32) -> 2 WMMAs per K-chunk.
__global__ __launch_bounds__(512)
void scan_rnn(const float* __restrict__ h0,
              const unsigned short* __restrict__ WhB,
              const unsigned short* __restrict__ xhB,
              const unsigned short* __restrict__ gateB,
              float* __restrict__ out) {
  __shared__ __align__(16) unsigned short sH[BB * LDA];
  __shared__ float sHf[BB * D];

  const int tid   = threadIdx.x;
  const int bbase = blockIdx.x * BB;

  for (int i = tid; i < BB * D; i += 512) {
    int r = i / D, c = i % D;
    float v = h0[(bbase + r) * D + c];
    sHf[r * D + c]  = v;
    sH[r * LDA + c] = f2bf(v);
  }
  __syncthreads();

  const int lane = tid & 31;
  const int wv   = tid >> 5;
  const int m16  = lane & 15;
  const int half = lane >> 4;
  const int n0   = wv * 32;

  float* h_all  = out;
  float* h_last = out + (long)B * T * D;
  const v8f zero = {0.f, 0.f, 0.f, 0.f, 0.f, 0.f, 0.f, 0.f};

  for (int t = 0; t < T; ++t) {
    v8f acc0 = zero, acc1 = zero;
    for (int k0 = 0; k0 < D; k0 += 32) {
      BfVec a;
      a.q[0] = *(const uint4*)(sH + m16 * LDA + k0 + half * 16);
      a.q[1] = *(const uint4*)(sH + m16 * LDA + k0 + half * 16 + 8);
      const unsigned short* p0 = WhB + (n0 + m16) * D + k0 + half * 16;
      const unsigned short* p1 = WhB + (n0 + 16 + m16) * D + k0 + half * 16;
      BfVec b0, b1;
      b0.q[0] = *(const uint4*)(p0);
      b0.q[1] = *(const uint4*)(p0 + 8);
      b1.q[0] = *(const uint4*)(p1);
      b1.q[1] = *(const uint4*)(p1 + 8);
      acc0 = __builtin_amdgcn_wmma_f32_16x16x32_bf16(false, a.v, false, b0.v,
                                                     (short)0, acc0, false, false);
      acc1 = __builtin_amdgcn_wmma_f32_16x16x32_bf16(false, a.v, false, b1.v,
                                                     (short)0, acc1, false, false);
    }
    __syncthreads();   // all waves done reading sH for this step

#pragma unroll
    for (int tt = 0; tt < 2; ++tt) {
      const v8f acc = tt ? acc1 : acc0;
      const int n = n0 + tt * 16 + m16;
#pragma unroll
      for (int j = 0; j < 8; ++j) {
        const int  mr  = j + 8 * half;            // local batch row
        const int  bg  = bbase + mr;              // global batch row
        const long idx = ((long)t * B + bg) * D + n;   // time-major ws
        const float xh = bf2f(xhB[idx]);
        const float g  = bf2f(gateB[idx]);
        const float hc = tanhf(acc[j] + xh);
        const float ho = sHf[mr * D + n];
        const float hn = g * hc + (1.0f - g) * ho;
        h_all[((long)bg * T + t) * D + n] = hn;   // [B,T,D] output
        sHf[mr * D + n]  = hn;
        sH[mr * LDA + n] = f2bf(hn);
        if (t == T - 1) h_last[(long)bg * D + n] = hn;
      }
    }
    __syncthreads();   // sH fully updated before next step's reads
  }
}

extern "C" void kernel_launch(void* const* d_in, const int* in_sizes, int n_in,
                              void* d_out, int out_size, void* d_ws, size_t ws_size,
                              hipStream_t stream) {
  // input order: x_seq, h0, W_h, W_x, b_h, W_g, b_g
  const float* x  = (const float*)d_in[0];
  const float* h0 = (const float*)d_in[1];
  const float* Wh = (const float*)d_in[2];
  const float* Wx = (const float*)d_in[3];
  const float* bh = (const float*)d_in[4];
  const float* Wg = (const float*)d_in[5];
  const float* bg = (const float*)d_in[6];
  float* out = (float*)d_out;

  unsigned short* WhB   = (unsigned short*)d_ws;
  unsigned short* WxB   = WhB + (size_t)D * D;
  unsigned short* WgB   = WxB + (size_t)D * D;
  unsigned short* xhB   = WgB + (size_t)D * D;
  unsigned short* gateB = xhB + (size_t)B * T * D;

  cvt_weights<<<(D * D + 255) / 256, 256, 0, stream>>>(Wh, Wx, Wg, WhB, WxB, WgB);
  proj_gemm<<<(B * T) / 16, 256, 0, stream>>>(x, WxB, WgB, bh, bg, xhB, gateB);
  scan_rnn<<<B / BB, 512, 0, stream>>>(h0, WhB, xhB, gateB, out);
}
